// SAGE_FULL_DGL_38225208934548
// MI455X (gfx1250) — compile-verified, matrix operand order checked by reading, not run on hardware
//
#include <hip/hip_runtime.h>
#include <hip/hip_bf16.h>

typedef __attribute__((ext_vector_type(16))) __bf16        v16bf;
typedef __attribute__((ext_vector_type(2)))  __bf16        v2bf;
typedef __attribute__((ext_vector_type(8)))  float         v8f;
typedef __attribute__((ext_vector_type(8)))  unsigned int  v8u;

#define D_FEAT 128  // K per half (D_IN == D_HID == 128)

// ---------------------------------------------------------------------------
// zero fill (float4 grid-stride)
// ---------------------------------------------------------------------------
__global__ void zero_f4_kernel(float4* __restrict__ p, long long n4) {
    long long i = (long long)blockIdx.x * blockDim.x + threadIdx.x;
    long long stride = (long long)gridDim.x * blockDim.x;
    float4 z; z.x = 0.f; z.y = 0.f; z.z = 0.f; z.w = 0.f;
    for (; i < n4; i += stride) p[i] = z;
}

// ---------------------------------------------------------------------------
// in-degree (f32 so it feeds the reciprocal directly)
// ---------------------------------------------------------------------------
__global__ void degree_kernel(const int* __restrict__ dst, float* __restrict__ deg, int E) {
    int e = blockIdx.x * blockDim.x + threadIdx.x;
    if (e < E) atomicAdd(&deg[dst[e]], 1.0f);
}

__global__ void invdeg_kernel(float* __restrict__ deg, int N) {
    int v = blockIdx.x * blockDim.x + threadIdx.x;
    if (v < N) deg[v] = 1.0f / fmaxf(deg[v], 1.0f);
}

// ---------------------------------------------------------------------------
// edge scatter-add: one wave32 per edge; lane covers 4 consecutive floats.
// Feature matrix (51 MB) is L2-resident on MI455X (192 MB L2), so the gather
// and the f32 atomics mostly stay on-die.
// ---------------------------------------------------------------------------
__global__ void scatter_add_kernel(const float* __restrict__ feat,
                                   const int* __restrict__ src,
                                   const int* __restrict__ dst,
                                   float* __restrict__ agg, int E) {
    const int wpb = blockDim.x >> 5;                 // waves per block
    const int e   = blockIdx.x * wpb + (threadIdx.x >> 5);
    const int l   = threadIdx.x & 31;
    if (e >= E) return;

    // lookahead prefetch of a future edge's source row (global_prefetch_b8)
    const int ep = e + 4096;
    if (ep < E) __builtin_prefetch(feat + (size_t)src[ep] * D_FEAT + l * 4, 0, 0);

    const float4 v = *(const float4*)(feat + (size_t)src[e] * D_FEAT + l * 4);
    float* o = agg + (size_t)dst[e] * D_FEAT + l * 4;
    atomicAdd(o + 0, v.x);
    atomicAdd(o + 1, v.y);
    atomicAdd(o + 2, v.z);
    atomicAdd(o + 3, v.w);
}

// ---------------------------------------------------------------------------
// K offset within a 32-wide chunk for A/B 16-bit WMMA lane layout
// (ISA 7.12.2: VGPR pair j holds K = 2j (+8 for j>=4) + 8*laneGroup)
// ---------------------------------------------------------------------------
__device__ __forceinline__ int wmma_kk(int j, int g) {
    return (j < 4) ? (2 * j + 8 * g) : (16 + 2 * (j - 4) + 8 * g);
}

// ---------------------------------------------------------------------------
// Pre-pack [W_self ; W_neigh] (f32, row-major KxDOUT) into bf16 in the exact
// per-lane WMMA B order: slot s = (ct*8 + kc)*32 + lane holds 16 bf16
// (8 dwords). One thread per slot. Total: (DOUT/16)*8*32 slots.
// ---------------------------------------------------------------------------
__global__ void pack_w_kernel(const float* __restrict__ Wself,
                              const float* __restrict__ Wneigh,
                              unsigned int* __restrict__ outp, int dout) {
    const int slots = (dout / 16) * 8 * 32;
    const int s = blockIdx.x * blockDim.x + threadIdx.x;
    if (s >= slots) return;
    const int l  = s & 31;
    const int kc = (s >> 5) & 7;
    const int ct = s >> 8;
    const int g = l >> 4, m = l & 15;
    const int n = ct * 16 + m;
    const float* __restrict__ src = (kc < 4) ? Wself : Wneigh;
    const int halfOff = (kc & 3) * 32;
#pragma unroll
    for (int j = 0; j < 8; ++j) {
        const int k = halfOff + wmma_kk(j, g);
        v2bf p;
        p[0] = (__bf16)src[(size_t)k * dout + n];
        p[1] = (__bf16)src[(size_t)(k + 1) * dout + n];
        outp[(size_t)s * 8 + j] = __builtin_bit_cast(unsigned int, p);
    }
}

// ---------------------------------------------------------------------------
// Fused SAGE layer GEMM via v_wmma_f32_16x16x32_bf16:
//   out[v,:] = relu?( self[v,:] @ Wself + (agg[v,:]*invdeg[v]) @ Wneigh + b )
// Concatenated K = 256 (8 chunks of 32; kc<4 -> self half, kc>=4 -> mean half).
// Phase 1: block cooperatively stages the 16x256 A tile into LDS, already
//          scaled, bf16-converted, and swizzled into WMMA per-lane order.
// Phase 2: one wave per 16x16 output tile (wave id = column tile); A from LDS
//          (2x ds_load_b128), B from the pre-packed bf16 buffer
//          (2x global_load_b128), 8 WMMAs, bias(+relu), store.
// ---------------------------------------------------------------------------
template <int DOUT, bool RELU>
__global__ void sage_gemm_wmma_kernel(const float* __restrict__ selfF,
                                      const float* __restrict__ aggF,
                                      const float* __restrict__ invdeg,
                                      const unsigned int* __restrict__ bpk,
                                      const float* __restrict__ bias,
                                      float* __restrict__ outp, int nNodes) {
    static_assert(DOUT % 16 == 0, "DOUT tile");
    __shared__ __align__(16) uint4 ldsA[8][2][32];   // [kc][half][lane] = 8 KB

    const int rowBase = blockIdx.x * 16;

    // ---- phase 1: cooperative A-tile staging (256 slots) ----
    for (int s = threadIdx.x; s < 256; s += blockDim.x) {
        const int l  = s & 31;
        const int kc = s >> 5;
        const int g = l >> 4, m = l & 15;
        int row = rowBase + m;
        if (row >= nNodes) row = nNodes - 1;         // clamp: EXEC stays all-1s
        const bool selfHalf = (kc < 4);
        const float* __restrict__ src = selfHalf ? selfF : aggF;
        const float scale = selfHalf ? 1.0f : invdeg[row];
        const int halfOff = (kc & 3) * 32;
        unsigned int dw[8];
#pragma unroll
        for (int j = 0; j < 8; ++j) {
            const int k = halfOff + wmma_kk(j, g);
            const float2 av = *(const float2*)(src + (size_t)row * D_FEAT + k);
            v2bf p;
            p[0] = (__bf16)(av.x * scale);
            p[1] = (__bf16)(av.y * scale);
            dw[j] = __builtin_bit_cast(unsigned int, p);
        }
        uint4 lo; lo.x = dw[0]; lo.y = dw[1]; lo.z = dw[2]; lo.w = dw[3];
        uint4 hi; hi.x = dw[4]; hi.y = dw[5]; hi.z = dw[6]; hi.w = dw[7];
        ldsA[kc][0][l] = lo;
        ldsA[kc][1][l] = hi;
    }
    __syncthreads();

    // ---- phase 2: per-wave WMMA over its column tile ----
    const int ct = threadIdx.x >> 5;                 // wave id -> column tile
    const int l  = threadIdx.x & 31;
    const int g = l >> 4, m = l & 15;
    const int n = ct * 16 + m;

    v8f c = {};
#pragma unroll
    for (int kc = 0; kc < 8; ++kc) {
        const uint4 alo = ldsA[kc][0][l];
        const uint4 ahi = ldsA[kc][1][l];
        v8u au = {alo.x, alo.y, alo.z, alo.w, ahi.x, ahi.y, ahi.z, ahi.w};
        const v16bf a = __builtin_bit_cast(v16bf, au);

        const uint4* __restrict__ bp4 =
            (const uint4*)(bpk + ((size_t)(ct * 8 + kc) * 32 + l) * 8);
        const uint4 blo = bp4[0];
        const uint4 bhi = bp4[1];
        v8u bu = {blo.x, blo.y, blo.z, blo.w, bhi.x, bhi.y, bhi.z, bhi.w};
        const v16bf b = __builtin_bit_cast(v16bf, bu);

        c = __builtin_amdgcn_wmma_f32_16x16x32_bf16(
                /*neg_a=*/false, a, /*neg_b=*/false, b,
                /*c_mod=*/(short)0, c, /*reuse_a=*/false, /*reuse_b=*/false);
    }

    const float bv = bias[n];
#pragma unroll
    for (int r = 0; r < 8; ++r) {
        const int orow = rowBase + r + 8 * g;        // C/D: VGPR r -> M = r + 8*laneGroup
        float val = c[r] + bv;
        if (RELU) val = fmaxf(val, 0.0f);
        if (orow < nNodes) outp[(size_t)orow * DOUT + n] = val;
    }
}

// ---------------------------------------------------------------------------
// launcher
// ---------------------------------------------------------------------------
extern "C" void kernel_launch(void* const* d_in, const int* in_sizes, int n_in,
                              void* d_out, int out_size, void* d_ws, size_t ws_size,
                              hipStream_t stream) {
    const float* x   = (const float*)d_in[0];
    const int*   src = (const int*)d_in[1];
    const int*   dst = (const int*)d_in[2];
    const float* Ws1 = (const float*)d_in[3];
    const float* Wn1 = (const float*)d_in[4];
    const float* b1  = (const float*)d_in[5];
    const float* Ws2 = (const float*)d_in[6];
    const float* Wn2 = (const float*)d_in[7];
    const float* b2  = (const float*)d_in[8];

    const int N = in_sizes[0] / D_FEAT;
    const int E = in_sizes[1];

    // workspace: deg[Npad] | agg[N*128] | h1[N*128] | Bpk1(16K dw) | Bpk2(8K dw)
    const size_t Npad = ((size_t)N + 255) & ~(size_t)255;
    float* deg = (float*)d_ws;
    float* agg = deg + Npad;
    float* h1  = agg + (size_t)N * D_FEAT;
    unsigned int* bpk1 = (unsigned int*)(h1 + (size_t)N * D_FEAT);   // 8ct*8kc*32l*8dw
    unsigned int* bpk2 = bpk1 + 8 * 8 * 32 * 8;                      // 4ct*8kc*32l*8dw
    float* out = (float*)d_out;

    const long long aggN4 = (long long)N * D_FEAT / 4;
    const int zeroBlocks = 2048;

    // 0) pack weights into WMMA-order bf16 (tiny, L2-resident afterwards)
    pack_w_kernel<<<(8 * 8 * 32 + 255) / 256, 256, 0, stream>>>(Ws1, Wn1, bpk1, 128);
    pack_w_kernel<<<(4 * 8 * 32 + 255) / 256, 256, 0, stream>>>(Ws2, Wn2, bpk2, 64);

    // 1) zero deg + agg
    zero_f4_kernel<<<32, 256, 0, stream>>>((float4*)deg, (long long)Npad / 4);
    zero_f4_kernel<<<zeroBlocks, 256, 0, stream>>>((float4*)agg, aggN4);

    // 2) degree, 3) invdeg (in place)
    degree_kernel<<<(E + 255) / 256, 256, 0, stream>>>(dst, deg, E);
    invdeg_kernel<<<(N + 255) / 256, 256, 0, stream>>>(deg, N);

    // 4) layer-1 neighbor sum
    scatter_add_kernel<<<(E + 7) / 8, 256, 0, stream>>>(x, src, dst, agg, E);

    // 5) layer-1 fused GEMM + bias + relu (8 col tiles -> 256 threads)
    const int rowTiles = (N + 15) / 16;
    sage_gemm_wmma_kernel<128, true>
        <<<rowTiles, 256, 0, stream>>>(x, agg, deg, bpk1, b1, h1, N);

    // 6) re-zero agg, layer-2 neighbor sum over h1
    zero_f4_kernel<<<zeroBlocks, 256, 0, stream>>>((float4*)agg, aggN4);
    scatter_add_kernel<<<(E + 7) / 8, 256, 0, stream>>>(h1, src, dst, agg, E);

    // 7) layer-2 fused GEMM + bias (4 col tiles -> 128 threads)
    sage_gemm_wmma_kernel<64, false>
        <<<rowTiles, 128, 0, stream>>>(h1, agg, deg, bpk2, b2, out, N);
}
